// JointNetwork_21492016349402
// MI455X (gfx1250) — matrix operand order repaired and checked by reading, not compiled, not run on hardware
//
#include <hip/hip_runtime.h>
#include <hip/hip_bf16.h>
#include <math.h>

// ---- types -----------------------------------------------------------------
typedef _Float16 half_t;
typedef __attribute__((ext_vector_type(16))) _Float16 v16h;
typedef __attribute__((ext_vector_type(8)))  _Float16 v8h;
typedef __attribute__((ext_vector_type(8)))  float    v8f;
typedef __attribute__((ext_vector_type(4)))  float    v4f;

// Problem constants (fixed by reference)
#define BB 4
#define TT 256
#define UU 64
#define DE 512
#define DM 512
#define VV 1024
#define C2 1024   // 2*DM

// ---- prep: f32 -> f16 cast -------------------------------------------------
__global__ void k_cast_f16(const float* __restrict__ src, half_t* __restrict__ dst, int n) {
    int i = blockIdx.x * blockDim.x + threadIdx.x;
    if (i < n) dst[i] = (half_t)src[i];
}

// ---- prep: f32 [rows][cols] -> f16 transposed [cols][rows] -----------------
__global__ void k_transpose_f16(const float* __restrict__ src, half_t* __restrict__ dst,
                                int rows, int cols) {
    int r = blockIdx.x * blockDim.x + threadIdx.x;   // coalesced writes along r
    int c = blockIdx.y;
    if (r < rows) dst[(size_t)c * rows + r] = (half_t)src[(size_t)r * cols + c];
}

// ---- WMMA GEMM: C[M,N] = A[M,K] * B[K,N] (+bias) (tanh?) -------------------
// A: f16 row-major [M][lda].  BT: f16, BT[n][k] = B[k][n], row length ldb,
// with column offset k0 (so B element (k,n) = BT[n*ldb + k0 + k]).
// One wave32 computes one 16x16 tile via v_wmma_f32_16x16x32_f16.
template <bool TANH, bool OUT_F32>
__global__ void k_gemm_wmma(const half_t* __restrict__ A, int lda,
                            const half_t* __restrict__ BT, int ldb, int k0,
                            const float* __restrict__ bias,
                            void* __restrict__ Cout, int ldc,
                            int M, int N, int K) {
    const int lane = threadIdx.x & 31;
    const int wv   = threadIdx.x >> 5;
    const int tile = blockIdx.x * (blockDim.x >> 5) + wv;
    const int ntn  = N >> 4;
    const int mt   = tile / ntn;
    const int nt   = tile - mt * ntn;
    if (mt * 16 >= M) return;

    const int idx  = lane & 15;   // M-row for A frag, N-col for B frag / C
    const int hsel = lane >> 4;

    // A frag (16x32 f16): lane holds row idx, K = hsel*8+{0..7} and 16+hsel*8+{0..7}
    const half_t* aptr = A + (size_t)(mt * 16 + idx) * lda + hsel * 8;
    // B frag (32x16 f16): lane holds col idx, K = hsel*8+{0..7} and 16+hsel*8+{0..7}
    const half_t* bptr = BT + (size_t)(nt * 16 + idx) * ldb + k0 + hsel * 8;

    const float bval = bias ? bias[nt * 16 + idx] : 0.0f;   // C layout: N = lane&15
    v8f acc = {bval, bval, bval, bval, bval, bval, bval, bval};

    for (int kk = 0; kk < K; kk += 32) {
        v8h alo = *(const v8h*)(aptr + kk);
        v8h ahi = *(const v8h*)(aptr + kk + 16);
        v8h blo = *(const v8h*)(bptr + kk);
        v8h bhi = *(const v8h*)(bptr + kk + 16);
        v16h a = __builtin_shufflevector(alo, ahi, 0,1,2,3,4,5,6,7,8,9,10,11,12,13,14,15);
        v16h b = __builtin_shufflevector(blo, bhi, 0,1,2,3,4,5,6,7,8,9,10,11,12,13,14,15);
        acc = __builtin_amdgcn_wmma_f32_16x16x32_f16(
            /*neg_a=*/false, a, /*neg_b=*/false, b,
            /*c_mod=*/(short)0, acc, /*reuse_a=*/false, /*reuse_b=*/false);
    }

    // C/D layout: VGPR r -> M = mt*16 + hsel*8 + r, N = nt*16 + idx
    if (OUT_F32) {
        float* c = (float*)Cout;
#pragma unroll
        for (int r = 0; r < 8; ++r) {
            int m = mt * 16 + hsel * 8 + r;
            c[(size_t)m * ldc + nt * 16 + idx] = acc[r];
        }
    } else {
        half_t* c = (half_t*)Cout;
#pragma unroll
        for (int r = 0; r < 8; ++r) {
            int m = mt * 16 + hsel * 8 + r;
            float v = TANH ? tanhf(acc[r]) : acc[r];
            c[(size_t)m * ldc + nt * 16 + idx] = (half_t)v;
        }
    }
}

// ---- final: out[b,t,u,:] = A1[b,t,:] + A2[b,u,:] + bj  (256 MB stream) -----
__global__ void k_joint_add(const float* __restrict__ A1, const float* __restrict__ A2,
                            const float* __restrict__ bj, float* __restrict__ out) {
    const int btu = blockIdx.x;          // B*T*U blocks
    const int u   = btu & (UU - 1);
    const int bt  = btu >> 6;            // b*T + t
    const int b   = bt >> 8;             // T = 256
    const v4f* a1 = (const v4f*)(A1 + (size_t)bt * VV);
    const v4f* a2 = (const v4f*)(A2 + (size_t)(b * UU + u) * VV);
    const v4f* bb = (const v4f*)bj;
    v4f* o = (v4f*)(out + (size_t)btu * VV);
    const int i = threadIdx.x;           // 256 threads * 4 floats = V = 1024
    v4f r = a1[i] + a2[i] + bb[i];
    __builtin_nontemporal_store(r, &o[i]);  // output > L2, keep it out of cache
}

// ---- launcher --------------------------------------------------------------
extern "C" void kernel_launch(void* const* d_in, const int* in_sizes, int n_in,
                              void* d_out, int out_size, void* d_ws, size_t ws_size,
                              hipStream_t stream) {
    const float* f  = (const float*)d_in[0];   // [B,T,DE]
    const float* g  = (const float*)d_in[1];   // [B,U,DD]
    const float* We = (const float*)d_in[2];   // [DE,DM]
    const float* be = (const float*)d_in[3];   // [DM]
    const float* Wd = (const float*)d_in[4];   // [DD,DM]
    const float* bd = (const float*)d_in[5];   // [DM]
    const float* Wj = (const float*)d_in[6];   // [2DM,V]
    const float* bj = (const float*)d_in[7];   // [V]
    float* out = (float*)d_out;

    // workspace carve (all sizes 256B-multiples)
    char* w = (char*)d_ws;
    half_t* fh  = (half_t*)w; w += (size_t)BB*TT*DE*2;   // 1 MB
    half_t* gh  = (half_t*)w; w += (size_t)BB*UU*DE*2;   // 256 KB
    half_t* WeT = (half_t*)w; w += (size_t)DM*DE*2;      // 512 KB  WeT[n][k]
    half_t* WdT = (half_t*)w; w += (size_t)DM*DE*2;      // 512 KB
    half_t* WjT = (half_t*)w; w += (size_t)VV*C2*2;      // 2 MB    WjT[v][c]
    half_t* t1  = (half_t*)w; w += (size_t)BB*TT*DM*2;   // 1 MB    tanh(fe)
    half_t* t2  = (half_t*)w; w += (size_t)BB*UU*DM*2;   // 256 KB  tanh(gd)
    float*  A1  = (float*)w;  w += (size_t)BB*TT*VV*4;   // 4 MB
    float*  A2  = (float*)w;  w += (size_t)BB*UU*VV*4;   // 1 MB

    // 1) precision casts + weight transposes
    {
        int n = BB*TT*DE;
        k_cast_f16<<<(n + 255) / 256, 256, 0, stream>>>(f, fh, n);
        n = BB*UU*DE;
        k_cast_f16<<<(n + 255) / 256, 256, 0, stream>>>(g, gh, n);
    }
    k_transpose_f16<<<dim3(DE / 256, DM), 256, 0, stream>>>(We, WeT, DE, DM);
    k_transpose_f16<<<dim3(DE / 256, DM), 256, 0, stream>>>(Wd, WdT, DE, DM);
    k_transpose_f16<<<dim3(C2 / 256, VV), 256, 0, stream>>>(Wj, WjT, C2, VV);

    // 2) projections + tanh (fused):  t1 = tanh(f@We+be), t2 = tanh(g@Wd+bd)
    {   // M=1024,N=512 -> 2048 tiles / 8 waves = 256 blocks
        k_gemm_wmma<true, false><<<(BB*TT/16)*(DM/16)/8, 256, 0, stream>>>(
            fh, DE, WeT, DE, 0, be, t1, DM, BB*TT, DM, DE);
        k_gemm_wmma<true, false><<<(BB*UU/16)*(DM/16)/8, 256, 0, stream>>>(
            gh, DE, WdT, DE, 0, bd, t2, DM, BB*UU, DM, DE);
    }

    // 3) factored joint GEMMs: A1 = t1 @ Wj[:DM], A2 = t2 @ Wj[DM:]
    k_gemm_wmma<false, true><<<(BB*TT/16)*(VV/16)/8, 256, 0, stream>>>(
        t1, DM, WjT, C2, 0, nullptr, A1, VV, BB*TT, VV, DM);
    k_gemm_wmma<false, true><<<(BB*UU/16)*(VV/16)/8, 256, 0, stream>>>(
        t2, DM, WjT, C2, DM, nullptr, A2, VV, BB*UU, VV, DM);

    // 4) bandwidth-bound broadcast add: out[b,t,u,:] = A1 + A2 + bj
    k_joint_add<<<BB*TT*UU, 256, 0, stream>>>(A1, A2, bj, out);
}
